// GradientConditionedAttention_74577812127915
// MI455X (gfx1250) — matrix operand order repaired and checked by reading, not compiled, not run on hardware
//
#include <hip/hip_runtime.h>
#include <hip/hip_bf16.h>
#include <math.h>

// ---------------------------------------------------------------------------
// GradientConditionedAttention for MI455X (gfx1250, wave32)
//   D=1024, H=16, hd=64, S=1024, B=4
// All matmuls via v_wmma_f32_16x16x32_bf16 (f32 accum).
// Attention K/V tiles staged to LDS via TDM tensor_load_to_lds (TENSORcnt),
// double-buffered; GEMM K-loops software-pipelined in registers.
// ---------------------------------------------------------------------------

#define D_MODEL 1024
#define N_HEADS 16
#define SEQ     1024
#define BATCH   4
#define HD      64
#define TOKENS  (BATCH * SEQ)   // 4096

typedef __bf16 bf16;
typedef __attribute__((ext_vector_type(16))) __bf16 v16bf;
typedef __attribute__((ext_vector_type(8)))  float  v8f;
typedef __attribute__((ext_vector_type(4)))  unsigned int v4u;
typedef __attribute__((ext_vector_type(8)))  int v8i;
typedef __attribute__((ext_vector_type(4)))  int v4i;

#if __has_builtin(__builtin_amdgcn_tensor_load_to_lds) && \
    __has_builtin(__builtin_amdgcn_s_wait_tensorcnt)
#define USE_TDM 1
#else
#define USE_TDM 0
#endif

union FragBF {            // one 16x16x32 bf16 A/B fragment: 16 bf16 per lane
    v16bf v;
    uint4 u[2];
    bf16  h[16];
};

__device__ __forceinline__ v8f vzero8() {
    v8f z;
#pragma unroll
    for (int i = 0; i < 8; ++i) z[i] = 0.0f;
    return z;
}

// cross-lane xor shuffle within the 16-lane half groups (wave32)
__device__ __forceinline__ float lane_xor(float x, int m) {
    int lane = threadIdx.x & 31;
    int src  = lane ^ m;
    return __int_as_float(
        __builtin_amdgcn_ds_bpermute(src << 2, __float_as_int(x)));
}

__device__ __forceinline__ v8f wmma_bf16(const FragBF& a, const FragBF& b, v8f c) {
    return __builtin_amdgcn_wmma_f32_16x16x32_bf16(
        false, a.v, false, b.v, (short)0, c, false, false);
}

#if USE_TDM
// byte offset of a __shared__ object inside the workgroup LDS allocation
__device__ __forceinline__ unsigned lds_off_u32(const void* p) {
    return (unsigned)(unsigned long long)(uintptr_t)
        (const __attribute__((address_space(3))) char*)p;
}

// Issue one 2D TDM tile load: dim0 = contiguous bf16 elems per row,
// dim1 = rows, stride0 = row stride in elems.  D# per ISA 08_async_tensor §8.
__device__ __forceinline__ void tdm_load_2d(const void* gaddr, unsigned lds_off,
                                            unsigned dim0, unsigned dim1,
                                            unsigned stride0) {
    unsigned long long ga = (unsigned long long)(uintptr_t)gaddr;
    v4u g0;
    g0[0] = 1u;                                    // count=1, user descriptor
    g0[1] = lds_off;                               // lds_addr [63:32]
    g0[2] = (unsigned)(ga & 0xffffffffu);          // global_addr[31:0]
    g0[3] = (unsigned)((ga >> 32) & 0x01ffffffu)   // global_addr[56:32]
          | 0x80000000u;                           // type=2 (bits 127:126=10b)
    v8i g1;
    g1[0] = 0x00010000;                            // data_size=1 (2B), mask=0
    g1[1] = (int)((dim0 & 0xffffu) << 16);         // tensor_dim0[15:0]
    g1[2] = (int)(((dim0 >> 16) & 0xffffu) | ((dim1 & 0xffffu) << 16));
    g1[3] = (int)(((dim1 >> 16) & 0xffffu) | ((dim0 & 0xffffu) << 16)); // tile_dim0
    g1[4] = (int)(dim1 & 0xffffu);                 // tile_dim1 (tile_dim2=0)
    g1[5] = (int)stride0;                          // tensor_dim0_stride[31:0]
    g1[6] = 0;                                     // stride hi / dim1_stride lo
    g1[7] = 0;
    v4i z4 = {0, 0, 0, 0};
    v8i z8 = {0, 0, 0, 0, 0, 0, 0, 0};
    // 6-arg form (clang-23 / therock-10.0 headers)
    __builtin_amdgcn_tensor_load_to_lds(g0, g1, z4, z4, z8, 0);
}
#endif

// ---------------------------------------------------------------------------
// Kernel 1: f32 -> bf16 conversion (vectorized x4)
// ---------------------------------------------------------------------------
__global__ void cvt_f32_to_bf16(const float* __restrict__ src,
                                bf16* __restrict__ dst, int n4) {
    int i = blockIdx.x * blockDim.x + threadIdx.x;
    if (i < n4) {
        float4 f = ((const float4*)src)[i];
        union { bf16 h[4]; uint2 u; } p;
        p.h[0] = (bf16)f.x; p.h[1] = (bf16)f.y;
        p.h[2] = (bf16)f.z; p.h[3] = (bf16)f.w;
        ((uint2*)dst)[i] = p.u;
    }
}

// ---------------------------------------------------------------------------
// Kernel 2: importance vectors  kimp[h][s] = dot(Wimp[h], ema[s]) + bimp[h]
// ---------------------------------------------------------------------------
__global__ void imp_kernel(const float* __restrict__ Wimp,
                           const float* __restrict__ bimp,
                           const float* __restrict__ k_ema,
                           const float* __restrict__ v_ema,
                           float* __restrict__ kimp,
                           float* __restrict__ vimp) {
    int idx = blockIdx.x * blockDim.x + threadIdx.x;   // 2*16*1024
    int sel = idx >> 14;
    int rem = idx & 16383;
    int h = rem >> 10;
    int s = rem & 1023;
    const float* w = Wimp + h * D_MODEL;
    const float* e = (sel ? v_ema : k_ema) + s * D_MODEL;
    float acc = 0.0f;
    for (int k = 0; k < D_MODEL; k += 4) {
        float4 a = *(const float4*)(w + k);
        float4 b = *(const float4*)(e + k);
        acc += a.x * b.x + a.y * b.y + a.z * b.z + a.w * b.w;
    }
    acc += bimp[h];
    (sel ? vimp : kimp)[h * SEQ + s] = acc;
}

// ---------------------------------------------------------------------------
// Kernel 3: fused QKV projection (software-pipelined K-loop).
//   C[4096, 3072] = x_bf @ [Wq|Wk|Wv]^T  (+ biases)
//   Q,K stored [B,H,S,hd] bf16;  V stored transposed [B,H,hd,S] bf16.
// ---------------------------------------------------------------------------
__global__ __launch_bounds__(128)
void qkv_proj_kernel(const bf16* __restrict__ xb,
                     const bf16* __restrict__ wqkv,
                     const float* __restrict__ bq,
                     const float* __restrict__ bk,
                     const float* __restrict__ bv,
                     bf16* __restrict__ qb,
                     bf16* __restrict__ kb,
                     bf16* __restrict__ vtb) {
    const int lane = threadIdx.x & 31;
    const int wid  = blockIdx.x * (blockDim.x >> 5) + (threadIdx.x >> 5);
    const int mt = wid / 48;
    const int ns = wid % 48;
    const int m0 = mt * 16;
    const int n0 = ns * 64;
    const int hi = lane >> 4, lo = lane & 15;

    const bf16* xrow = xb + (size_t)(m0 + lo) * D_MODEL;

    v8f acc[4];
#pragma unroll
    for (int t = 0; t < 4; ++t) acc[t] = vzero8();

    FragBF a_c, b_c[4];
    a_c.u[0] = *(const uint4*)(xrow + 16 * hi);
    a_c.u[1] = *(const uint4*)(xrow + 16 * hi + 8);
#pragma unroll
    for (int nt = 0; nt < 4; ++nt) {
        const bf16* wr = wqkv + (size_t)(n0 + nt * 16 + lo) * D_MODEL + 16 * hi;
        b_c[nt].u[0] = *(const uint4*)(wr);
        b_c[nt].u[1] = *(const uint4*)(wr + 8);
    }

    for (int kk = 0; kk < D_MODEL - 32; kk += 32) {
        const int kn = kk + 32;
        FragBF a_n, b_n[4];
        a_n.u[0] = *(const uint4*)(xrow + kn + 16 * hi);
        a_n.u[1] = *(const uint4*)(xrow + kn + 16 * hi + 8);
#pragma unroll
        for (int nt = 0; nt < 4; ++nt) {
            const bf16* wr = wqkv + (size_t)(n0 + nt * 16 + lo) * D_MODEL + kn + 16 * hi;
            b_n[nt].u[0] = *(const uint4*)(wr);
            b_n[nt].u[1] = *(const uint4*)(wr + 8);
        }
#pragma unroll
        for (int nt = 0; nt < 4; ++nt)
            acc[nt] = wmma_bf16(a_c, b_c[nt], acc[nt]);
        a_c = a_n;
#pragma unroll
        for (int nt = 0; nt < 4; ++nt) b_c[nt] = b_n[nt];
    }
#pragma unroll
    for (int nt = 0; nt < 4; ++nt)
        acc[nt] = wmma_bf16(a_c, b_c[nt], acc[nt]);

#pragma unroll
    for (int nt = 0; nt < 4; ++nt) {
        int n   = n0 + nt * 16 + lo;      // 0..3071
        int sec = n >> 10;                // 0=Q, 1=K, 2=V
        int nf  = n & 1023;
        int h   = nf >> 6;
        int d   = nf & 63;
        float bias = (sec == 0) ? bq[nf] : (sec == 1) ? bk[nf] : bv[nf];
        if (sec < 2) {
            bf16* base = (sec == 0) ? qb : kb;
#pragma unroll
            for (int r = 0; r < 8; ++r) {
                int tok = m0 + r + 8 * hi;
                int b_  = tok >> 10;
                int s   = tok & 1023;
                base[((size_t)(b_ * N_HEADS + h) * SEQ + s) * HD + d] =
                    (bf16)(acc[nt][r] + bias);
            }
        } else {
            int tok0 = m0 + 8 * hi;
            int b_   = tok0 >> 10;
            int s0   = tok0 & 1023;
            union { bf16 h8[8]; uint4 u; } pk;
#pragma unroll
            for (int r = 0; r < 8; ++r) pk.h8[r] = (bf16)(acc[nt][r] + bias);
            *(uint4*)(vtb + ((size_t)(b_ * N_HEADS + h) * HD + d) * SEQ + s0) = pk.u;
        }
    }
}

// ---------------------------------------------------------------------------
// Kernel 4: causal flash attention (one 16-query tile per wave, 2 waves/block)
//   K chunk (32x64) and V chunk (64x32, from Vt) staged into LDS by TDM,
//   double-buffered; P transposed C->A layout through LDS.
// ---------------------------------------------------------------------------
__global__ __launch_bounds__(64)
void attn_kernel(const bf16* __restrict__ qb,
                 const bf16* __restrict__ kb,
                 const bf16* __restrict__ vtb,
                 const float* __restrict__ kimp,
                 const float* __restrict__ vimp,
                 bf16* __restrict__ ob) {
#if USE_TDM
    __shared__ bf16 kbuf[2][2][32 * 64];   // [wave][buf][rows 32][d 64]
    __shared__ bf16 vbuf[2][2][64 * 32];   // [wave][buf][d 64][k 32]
#endif
    __shared__ bf16 pbuf[2][16 * 32];

    const int lane = threadIdx.x & 31;
    const int w    = threadIdx.x >> 5;
    const int wid  = blockIdx.x * 2 + w;
    const int qt = wid & 63;
    const int bh = wid >> 6;
    const int h  = bh & 15;
    const int b  = bh >> 4;
    const int q0 = qt * 16;
    const int hi = lane >> 4, lo = lane & 15;

    const bf16*  qbase  = qb  + (size_t)(b * N_HEADS + h) * SEQ * HD;
    const bf16*  kbase  = kb  + (size_t)(b * N_HEADS + h) * SEQ * HD;
    const bf16*  vbase  = vtb + (size_t)(b * N_HEADS + h) * HD * SEQ;
    const float* kimp_h = kimp + h * SEQ;
    const float* vimp_h = vimp + h * SEQ;

    FragBF a0, a1;
    {
        const bf16* qr = qbase + (size_t)(q0 + lo) * HD;
        a0.u[0] = *(const uint4*)(qr + 16 * hi);
        a0.u[1] = *(const uint4*)(qr + 16 * hi + 8);
        a1.u[0] = *(const uint4*)(qr + 32 + 16 * hi);
        a1.u[1] = *(const uint4*)(qr + 32 + 16 * hi + 8);
    }

    float mrow[8], lrow[8], kib[8];
    v8f o[4];
#pragma unroll
    for (int r = 0; r < 8; ++r) {
        mrow[r] = -3.0e38f;
        lrow[r] = 0.0f;
        kib[r]  = kimp_h[q0 + r + 8 * hi];
    }
#pragma unroll
    for (int t = 0; t < 4; ++t) o[t] = vzero8();

    const int nchunk = (q0 + 16 + 31) >> 5;     // causal truncation

#if USE_TDM
    // prologue: stage chunk 0
    tdm_load_2d(kbase, lds_off_u32(&kbuf[w][0][0]), 64, 32, 64);
    tdm_load_2d(vbase, lds_off_u32(&vbuf[w][0][0]), 32, 64, SEQ);
#endif

    for (int c = 0; c < nchunk; ++c) {
        const int kc  = c << 5;
        const int buf = c & 1;

#if USE_TDM
        if (c + 1 < nchunk) {
            // protect buf[(c+1)&1] against in-flight ds reads, then prefetch
            asm volatile("s_wait_dscnt 0" ::: "memory");
            tdm_load_2d(kbase + (size_t)(kc + 32) * HD,
                        lds_off_u32(&kbuf[w][buf ^ 1][0]), 64, 32, 64);
            tdm_load_2d(vbase + kc + 32,
                        lds_off_u32(&vbuf[w][buf ^ 1][0]), 32, 64, SEQ);
            __builtin_amdgcn_s_wait_tensorcnt(2);   // chunk c complete
        } else {
            __builtin_amdgcn_s_wait_tensorcnt(0);
        }
        asm volatile("" ::: "memory");
#endif

        // --- scores: two 16x16 key subtiles, hd=64 => 2 WMMAs each ---
        v8f   sc[2];
        float vib[2];
#pragma unroll
        for (int kt = 0; kt < 2; ++kt) {
            int kidx = kc + kt * 16 + lo;
#if USE_TDM
            const bf16* kr = &kbuf[w][buf][(kt * 16 + lo) * 64 + 16 * hi];
#else
            const bf16* kr = kbase + (size_t)kidx * HD + 16 * hi;
#endif
            FragBF b0, b1;
            b0.u[0] = *(const uint4*)(kr);
            b0.u[1] = *(const uint4*)(kr + 8);
            b1.u[0] = *(const uint4*)(kr + 32);
            b1.u[1] = *(const uint4*)(kr + 40);
            v8f z = vzero8();
            z = wmma_bf16(a0, b0, z);
            z = wmma_bf16(a1, b1, z);
            sc[kt]  = z;
            vib[kt] = vimp_h[kidx];
        }

        // --- bias + causal mask + per-tile row max ---
        float tmax[8];
#pragma unroll
        for (int r = 0; r < 8; ++r) {
            int m = q0 + r + 8 * hi;
            float s0v = sc[0][r] * 0.125f + 0.05f * (kib[r] + vib[0]);
            float s1v = sc[1][r] * 0.125f + 0.05f * (kib[r] + vib[1]);
            if (kc + lo      > m) s0v = -3.0e38f;
            if (kc + 16 + lo > m) s1v = -3.0e38f;
            sc[0][r] = s0v; sc[1][r] = s1v;
            tmax[r]  = fmaxf(s0v, s1v);
        }
#pragma unroll
        for (int msk = 1; msk < 16; msk <<= 1)
#pragma unroll
            for (int r = 0; r < 8; ++r)
                tmax[r] = fmaxf(tmax[r], lane_xor(tmax[r], msk));

        // --- online softmax update ---
        float corr[8], rs[8];
#pragma unroll
        for (int r = 0; r < 8; ++r) {
            float mnew = fmaxf(mrow[r], tmax[r]);
            corr[r] = __expf(mrow[r] - mnew);
            mrow[r] = mnew;
            float p0 = __expf(sc[0][r] - mnew);
            float p1 = __expf(sc[1][r] - mnew);
            sc[0][r] = p0; sc[1][r] = p1;
            rs[r] = p0 + p1;
        }
#pragma unroll
        for (int msk = 1; msk < 16; msk <<= 1)
#pragma unroll
            for (int r = 0; r < 8; ++r)
                rs[r] += lane_xor(rs[r], msk);
#pragma unroll
        for (int r = 0; r < 8; ++r) lrow[r] = lrow[r] * corr[r] + rs[r];
#pragma unroll
        for (int t = 0; t < 4; ++t)
#pragma unroll
            for (int r = 0; r < 8; ++r) o[t][r] *= corr[r];

        // --- transpose P (C-layout -> A-layout) through LDS, wave-local ---
        bf16* pl = pbuf[w];
#pragma unroll
        for (int kt = 0; kt < 2; ++kt) {
            int col = kt * 16 + lo;
#pragma unroll
            for (int r = 0; r < 8; ++r)
                pl[(8 * hi + r) * 32 + col] = (bf16)sc[kt][r];
        }
        __builtin_amdgcn_wave_barrier();
        asm volatile("s_wait_dscnt 0" ::: "memory");

        FragBF pf;
        {
            const bf16* pr = pl + lo * 32;
            pf.u[0] = *(const uint4*)(pr + 16 * hi);
            pf.u[1] = *(const uint4*)(pr + 16 * hi + 8);
        }
        __builtin_amdgcn_wave_barrier();
        asm volatile("s_wait_dscnt 0" ::: "memory");

        // --- O += P @ V  (4 N-tiles of 16 over hd=64) ---
#pragma unroll
        for (int dt = 0; dt < 4; ++dt) {
#if USE_TDM
            const bf16* vr = &vbuf[w][buf][(dt * 16 + lo) * 32 + 16 * hi];
#else
            const bf16* vr = vbase + (size_t)(dt * 16 + lo) * SEQ + kc + 16 * hi;
#endif
            FragBF vf;
            vf.u[0] = *(const uint4*)(vr);
            vf.u[1] = *(const uint4*)(vr + 8);
            o[dt] = wmma_bf16(pf, vf, o[dt]);
        }
    }

    // --- finalize: O /= l, store bf16 [B,S,D] with head interleave ---
#pragma unroll
    for (int r = 0; r < 8; ++r) lrow[r] = 1.0f / lrow[r];
#pragma unroll
    for (int dt = 0; dt < 4; ++dt) {
        int dcol = h * HD + dt * 16 + lo;
#pragma unroll
        for (int r = 0; r < 8; ++r) {
            int s = q0 + r + 8 * hi;
            ob[((size_t)b * SEQ + s) * D_MODEL + dcol] = (bf16)(o[dt][r] * lrow[r]);
        }
    }
}

// ---------------------------------------------------------------------------
// Kernel 5: output projection  out[4096,1024] = Obf @ Wo^T + bo  (f32 out)
// (software-pipelined like kernel 3)
// ---------------------------------------------------------------------------
__global__ __launch_bounds__(128)
void out_proj_kernel(const bf16* __restrict__ ob,
                     const bf16* __restrict__ wob,
                     const float* __restrict__ bo,
                     float* __restrict__ out) {
    const int lane = threadIdx.x & 31;
    const int wid  = blockIdx.x * 4 + (threadIdx.x >> 5);
    const int mt = wid >> 4;
    const int ns = wid & 15;
    const int m0 = mt * 16;
    const int n0 = ns * 64;
    const int hi = lane >> 4, lo = lane & 15;

    const bf16* xrow = ob + (size_t)(m0 + lo) * D_MODEL;

    v8f acc[4];
#pragma unroll
    for (int t = 0; t < 4; ++t) acc[t] = vzero8();

    FragBF a_c, b_c[4];
    a_c.u[0] = *(const uint4*)(xrow + 16 * hi);
    a_c.u[1] = *(const uint4*)(xrow + 16 * hi + 8);
#pragma unroll
    for (int nt = 0; nt < 4; ++nt) {
        const bf16* wr = wob + (size_t)(n0 + nt * 16 + lo) * D_MODEL + 16 * hi;
        b_c[nt].u[0] = *(const uint4*)(wr);
        b_c[nt].u[1] = *(const uint4*)(wr + 8);
    }

    for (int kk = 0; kk < D_MODEL - 32; kk += 32) {
        const int kn = kk + 32;
        FragBF a_n, b_n[4];
        a_n.u[0] = *(const uint4*)(xrow + kn + 16 * hi);
        a_n.u[1] = *(const uint4*)(xrow + kn + 16 * hi + 8);
#pragma unroll
        for (int nt = 0; nt < 4; ++nt) {
            const bf16* wr = wob + (size_t)(n0 + nt * 16 + lo) * D_MODEL + kn + 16 * hi;
            b_n[nt].u[0] = *(const uint4*)(wr);
            b_n[nt].u[1] = *(const uint4*)(wr + 8);
        }
#pragma unroll
        for (int nt = 0; nt < 4; ++nt)
            acc[nt] = wmma_bf16(a_c, b_c[nt], acc[nt]);
        a_c = a_n;
#pragma unroll
        for (int nt = 0; nt < 4; ++nt) b_c[nt] = b_n[nt];
    }
#pragma unroll
    for (int nt = 0; nt < 4; ++nt)
        acc[nt] = wmma_bf16(a_c, b_c[nt], acc[nt]);

#pragma unroll
    for (int nt = 0; nt < 4; ++nt) {
        int n = n0 + nt * 16 + lo;
        float bias = bo[n];
#pragma unroll
        for (int r = 0; r < 8; ++r) {
            int m = m0 + r + 8 * hi;
            out[(size_t)m * D_MODEL + n] = acc[nt][r] + bias;
        }
    }
}

// ---------------------------------------------------------------------------
// Host: workspace layout + launches (all on `stream`, graph-capture safe)
// ---------------------------------------------------------------------------
extern "C" void kernel_launch(void* const* d_in, const int* in_sizes, int n_in,
                              void* d_out, int out_size, void* d_ws, size_t ws_size,
                              hipStream_t stream) {
    const float* x     = (const float*)d_in[0];
    const float* Wq    = (const float*)d_in[1];
    const float* bq    = (const float*)d_in[2];
    const float* Wk    = (const float*)d_in[3];
    const float* bk    = (const float*)d_in[4];
    const float* Wv    = (const float*)d_in[5];
    const float* bv    = (const float*)d_in[6];
    const float* Wo    = (const float*)d_in[7];
    const float* bo    = (const float*)d_in[8];
    const float* Wimp  = (const float*)d_in[9];
    const float* bimp  = (const float*)d_in[10];
    const float* k_ema = (const float*)d_in[11];
    const float* v_ema = (const float*)d_in[12];
    float* out = (float*)d_out;

    char* ws = (char*)d_ws;
    size_t off = 0;
    auto take = [&](size_t bytes) -> char* {
        char* p = ws + off;
        off = (off + bytes + 255) & ~(size_t)255;
        return p;
    };
    const size_t NB = (size_t)TOKENS * D_MODEL;        // 4M elems
    bf16*  xb    = (bf16*)take(NB * 2);
    bf16*  wqkvb = (bf16*)take(3 * (size_t)D_MODEL * D_MODEL * 2);
    bf16*  wob   = (bf16*)take((size_t)D_MODEL * D_MODEL * 2);
    bf16*  qb    = (bf16*)take(NB * 2);
    bf16*  kb    = (bf16*)take(NB * 2);
    bf16*  vtb   = (bf16*)take(NB * 2);
    bf16*  obuf  = (bf16*)take(NB * 2);
    float* kimp  = (float*)take((size_t)N_HEADS * SEQ * 4);
    float* vimp  = (float*)take((size_t)N_HEADS * SEQ * 4);

    // 1) conversions to bf16
    {
        int n4 = (int)(NB / 4);
        cvt_f32_to_bf16<<<n4 / 256, 256, 0, stream>>>(x, xb, n4);
        int w4 = (D_MODEL * D_MODEL) / 4;
        cvt_f32_to_bf16<<<w4 / 256, 256, 0, stream>>>(Wq, wqkvb, w4);
        cvt_f32_to_bf16<<<w4 / 256, 256, 0, stream>>>(Wk, wqkvb + (size_t)D_MODEL * D_MODEL, w4);
        cvt_f32_to_bf16<<<w4 / 256, 256, 0, stream>>>(Wv, wqkvb + 2 * (size_t)D_MODEL * D_MODEL, w4);
        cvt_f32_to_bf16<<<w4 / 256, 256, 0, stream>>>(Wo, wob, w4);
    }

    // 2) importance vectors
    imp_kernel<<<(2 * N_HEADS * SEQ) / 256, 256, 0, stream>>>(
        Wimp, bimp, k_ema, v_ema, kimp, vimp);

    // 3) fused QKV projection: 256 * 48 = 12288 waves, 4 waves/block
    qkv_proj_kernel<<<3072, 128, 0, stream>>>(xb, wqkvb, bq, bk, bv, qb, kb, vtb);

    // 4) flash attention: B*H*(S/16) = 4096 waves, 2 waves/block (LDS budget)
    attn_kernel<<<2048, 64, 0, stream>>>(qb, kb, vtb, kimp, vimp, obuf);

    // 5) output projection: 256 * 16 = 4096 waves, 4 waves/block
    out_proj_kernel<<<1024, 128, 0, stream>>>(obuf, wob, bo, out);
}
